// Bio_Network_21131239096984
// MI455X (gfx1250) — compile-verified, matrix operand order checked by reading, not run on hardware
//
#include <hip/hip_runtime.h>
#include <hip/hip_bf16.h>
#include <math.h>

#define NATOMS 192
#define ZB 16
#define NB 40
#define KPAD 64
#define HDIM 64

typedef __attribute__((ext_vector_type(16))) __bf16 v16bf;
typedef __attribute__((ext_vector_type(8)))  float  v8f;

union Frag { v16bf v; uint4 q[2]; };

__device__ __forceinline__ unsigned short f2bf(float f) {
  unsigned int u = __float_as_uint(f);
  u += 0x7FFFu + ((u >> 16) & 1u);           // round-to-nearest-even
  return (unsigned short)(u >> 16);
}

// fast softplus via hw transcendentals (v_exp_f32/v_log_f32); outputs feed
// bf16 storage, so fp32-exact log1p is unnecessary. For bx>15, log(1+e^bx)=bx
// to well below bf16 ulp.
__device__ __forceinline__ float softplus5(float x) {
  float bx = 5.0f * x;
  float s = __logf(1.0f + __expf(bx));
  return (bx > 15.0f) ? bx : s;
}
__device__ __forceinline__ float sspf(float x) {   // softplus(5x)/5 - ln2/5
  return softplus5(x) * 0.2f - 0.13862943611198906f;
}
__device__ __forceinline__ float spf(float x) {    // softplus(5x)/5
  return softplus5(x) * 0.2f;
}

// ---------------- small prep kernels ----------------

__global__ void k_encode(const float* __restrict__ feat,
                         const float* __restrict__ Wb, const float* __restrict__ bb,
                         const float* __restrict__ Wc, const float* __restrict__ bc,
                         float* __restrict__ fbio, float* __restrict__ fch) {
  int t = blockIdx.x * blockDim.x + threadIdx.x;
  if (t >= ZB * NATOMS * HDIM) return;
  int j = t & 63;
  int row = t >> 6;
  const float* fr = feat + row * 8;
  float acc = bb[j];
#pragma unroll
  for (int c = 0; c < 7; ++c) acc += fr[c] * Wb[c * HDIM + j];
  fbio[t] = acc;
  fch[t] = fr[7] * Wc[j] + bc[j];
}

__global__ void k_basis(const float* __restrict__ geom, unsigned short* __restrict__ basis) {
  int t = blockIdx.x * blockDim.x + threadIdx.x;
  if (t >= ZB * NATOMS * NATOMS) return;
  int b = t % NATOMS;
  int a = (t / NATOMS) % NATOMS;
  int z = t / (NATOMS * NATOMS);
  const float* ga = geom + (z * NATOMS + a) * 3;
  const float* gb = geom + (z * NATOMS + b) * 3;
  float dx = ga[0] - gb[0], dy = ga[1] - gb[1], dz = ga[2] - gb[2];
  float r = sqrtf(dx * dx + dy * dy + dz * dz);
  float rn = r * (39.0f / 10.0f);            // r/step, step = 10/39
  unsigned short* out = basis + (long)t * KPAD;
#pragma unroll
  for (int k8 = 0; k8 < KPAD; k8 += 8) {     // vectorized b128 stores
    unsigned int p[4];
#pragma unroll
    for (int i = 0; i < 4; ++i) {
      unsigned int lo = 0, hi = 0;
      int k = k8 + 2 * i;
      if (k < NB) {
        float d = rn - (float)k;
        if (fabsf(d) < 1.0f) { float c = __cosf(1.57079632679f * d); lo = f2bf(c * c); }
      }
      if (k + 1 < NB) {
        float d = rn - (float)(k + 1);
        if (fabsf(d) < 1.0f) { float c = __cosf(1.57079632679f * d); hi = f2bf(c * c); }
      }
      p[i] = lo | (hi << 16);
    }
    uint4 q; q.x = p[0]; q.y = p[1]; q.z = p[2]; q.w = p[3];
    *(uint4*)(out + k8) = q;
  }
}

// pack W[Ksrc,N] (f32) -> Bt[N,64] (bf16), zero-padded in K
__global__ void k_packT(const float* __restrict__ W, unsigned short* __restrict__ Bt,
                        int Ksrc, int N) {
  int t = blockIdx.x * blockDim.x + threadIdx.x;
  if (t >= N * 64) return;
  int n = t / 64, k = t % 64;
  Bt[t] = f2bf(k < Ksrc ? W[k * N + n] : 0.0f);
}

__global__ void k_cast_bf(const float* __restrict__ src, unsigned short* __restrict__ dst, long n) {
  long t = (long)blockIdx.x * blockDim.x + threadIdx.x;
  if (t < n) dst[t] = f2bf(src[t]);
}

__global__ void k_prep_fm(const float* __restrict__ f, const float* __restrict__ mask,
                          unsigned short* __restrict__ fm) {
  int t = blockIdx.x * blockDim.x + threadIdx.x;
  if (t >= ZB * NATOMS * HDIM) return;
  int row = t >> 6;
  fm[t] = f2bf(f[t] * mask[row] * 0.07216878364870323f);   // 1/sqrt(192)
}

__global__ void k_zero(float* __restrict__ p, int n) {
  int t = blockIdx.x * blockDim.x + threadIdx.x;
  if (t < n) p[t] = 0.0f;
}

__global__ void k_finalize(const float* __restrict__ acc, const float* __restrict__ mask,
                           float* __restrict__ f) {
  int t = blockIdx.x * blockDim.x + threadIdx.x;
  if (t >= ZB * NATOMS * HDIM) return;
  f[t] = spf(acc[t]) * mask[t >> 6];
}

// ---------------- fused radial MLP: h = ssp(ssp(basis@W1+b1)@W2+b2) ----------------
// 8 waves/block; each wave owns 16 rows x 64 cols. GEMM1 result goes through
// a wave-private LDS tile (DS ops in-order per wave -> no barrier), then is
// re-fragmented as the A operand of GEMM2. h1 never touches HBM.
__global__ __launch_bounds__(256)
void k_h_fused(const unsigned short* __restrict__ basis,
               const unsigned short* __restrict__ W1t, const float* __restrict__ b1,
               const unsigned short* __restrict__ W2t, const float* __restrict__ b2,
               unsigned short* __restrict__ hbuf) {
  __shared__ unsigned short lds[8][16 * 64];
  const int lane = threadIdx.x & 31;
  const int w = threadIdx.x >> 5;
  const long bm = (long)blockIdx.y * 128 + w * 16;
  const int r15 = lane & 15;
  const int khalf = lane >> 4;

  const unsigned short* arow = basis + (bm + r15) * 64;

  v8f acc[4] = {{}, {}, {}, {}};
#pragma unroll
  for (int kk = 0; kk < 64; kk += 32) {
    Frag a;
    const int k0 = kk + khalf * 8;
    a.q[0] = *(const uint4*)(arow + k0);
    a.q[1] = *(const uint4*)(arow + k0 + 16);
#pragma unroll
    for (int t = 0; t < 4; ++t) {
      Frag b;
      const unsigned short* brow = W1t + (t * 16 + r15) * 64;
      b.q[0] = *(const uint4*)(brow + k0);
      b.q[1] = *(const uint4*)(brow + k0 + 16);
      acc[t] = __builtin_amdgcn_wmma_f32_16x16x32_bf16(false, a.v, false, b.v,
                                                       (short)0, acc[t], false, false);
    }
  }
  // ssp + bias -> wave-private LDS tile [16 rows][64 cols]
#pragma unroll
  for (int t = 0; t < 4; ++t) {
    int n = t * 16 + r15;
#pragma unroll
    for (int r = 0; r < 8; ++r) {
      int m = r + 8 * khalf;
      lds[w][m * 64 + n] = f2bf(sspf(acc[t][r] + b1[n]));
    }
  }
  // GEMM2: A from LDS (same wave's 16 rows), B = W2t
  v8f acc2[4] = {{}, {}, {}, {}};
  const unsigned short* lrow = &lds[w][r15 * 64];
#pragma unroll
  for (int kk = 0; kk < 64; kk += 32) {
    Frag a;
    const int k0 = kk + khalf * 8;
    a.q[0] = *(const uint4*)(lrow + k0);
    a.q[1] = *(const uint4*)(lrow + k0 + 16);
#pragma unroll
    for (int t = 0; t < 4; ++t) {
      Frag b;
      const unsigned short* brow = W2t + (t * 16 + r15) * 64;
      b.q[0] = *(const uint4*)(brow + k0);
      b.q[1] = *(const uint4*)(brow + k0 + 16);
      acc2[t] = __builtin_amdgcn_wmma_f32_16x16x32_bf16(false, a.v, false, b.v,
                                                        (short)0, acc2[t], false, false);
    }
  }
#pragma unroll
  for (int t = 0; t < 4; ++t) {
    int n = t * 16 + r15;
#pragma unroll
    for (int r = 0; r < 8; ++r) {
      long m = bm + r + 8 * khalf;
      hbuf[m * 64 + n] = f2bf(sspf(acc2[t][r] + b2[n]));
    }
  }
}

// ---------------- generic bf16 WMMA GEMM (wave tile 16x64) ----------------
// A [rows,K] bf16; Bt [cols,K] bf16 (B transposed). Each wave: 16 rows x 64
// cols, one A fragment feeds 4 wmma per K-step (A streamed once from HBM).
// mode 1: G3 -> tmpT[z][j][b*64+h] scatter, raw bf16 (rows = z*192+b, n = h*64+j)
// mode 3: G4 k-split partial sums -> unsafeAtomicAdd into f32 accum [z,192,64]
//         blockIdx.z = z*KS + kchunk, kLen = K/KS
#define KS 4
__global__ void k_gemm2(const unsigned short* __restrict__ A,
                        const unsigned short* __restrict__ Bt,
                        void* __restrict__ Dst,
                        int K, int kLen,
                        long strideA, long strideB,
                        int zdiv, int mode) {
  const int nwaves = blockDim.x >> 5;
  const int lane = threadIdx.x & 31;
  const int w = threadIdx.x >> 5;
  const int r15 = lane & 15;
  const int khalf = lane >> 4;
  const int z = blockIdx.z / zdiv;
  const int kc = blockIdx.z % zdiv;
  const long kOff = (long)kc * kLen;
  const long bm = (long)blockIdx.y * (nwaves * 16) + w * 16;
  const int bn = blockIdx.x * 64;

  const unsigned short* arow = A + (long)z * strideA + (bm + r15) * K + kOff;
  const unsigned short* bbase = Bt + (long)z * strideB + kOff;

  v8f acc[4] = {{}, {}, {}, {}};
  for (int kk = 0; kk < kLen; kk += 32) {
    Frag a;
    const int k0 = kk + khalf * 8;
    a.q[0] = *(const uint4*)(arow + k0);
    a.q[1] = *(const uint4*)(arow + k0 + 16);
#pragma unroll
    for (int t = 0; t < 4; ++t) {
      Frag b;
      const unsigned short* brow = bbase + (long)(bn + t * 16 + r15) * K;
      b.q[0] = *(const uint4*)(brow + k0);
      b.q[1] = *(const uint4*)(brow + k0 + 16);
      acc[t] = __builtin_amdgcn_wmma_f32_16x16x32_bf16(false, a.v, false, b.v,
                                                       (short)0, acc[t], false, false);
    }
  }

#pragma unroll
  for (int t = 0; t < 4; ++t) {
    int n = bn + t * 16 + r15;
#pragma unroll
    for (int r = 0; r < 8; ++r) {
      long m = bm + r + 8 * khalf;
      float v = acc[t][r];
      if (mode == 1) {
        unsigned short* D = (unsigned short*)Dst;
        int zz = (int)(m / NATOMS), b = (int)(m % NATOMS);
        long idx = (long)zz * 64 * 12288 + (long)(n & 63) * 12288 + b * 64 + (n >> 6);
        D[idx] = f2bf(v);
      } else {
        float* D = (float*)Dst;
        unsafeAtomicAdd(&D[((long)z * NATOMS + m) * 64 + n], v);
      }
    }
  }
}

// ---------------- head ----------------

__global__ void k_head1(const float* __restrict__ fbio, const float* __restrict__ fch,
                        const float* __restrict__ W, const float* __restrict__ b,
                        float* __restrict__ y) {
  int t = blockIdx.x * blockDim.x + threadIdx.x;
  if (t >= ZB * NATOMS * 128) return;
  int n = t & 127;
  int row = t >> 7;
  float acc = b[n];
  const float* fb = fbio + (long)row * 64;
  const float* fc = fch + (long)row * 64;
  for (int k = 0; k < 64; ++k) acc += fb[k] * W[k * 128 + n];
  for (int k = 0; k < 64; ++k) acc += fc[k] * W[(64 + k) * 128 + n];
  y[t] = acc;
}

__global__ void k_bn_lrelu(float* __restrict__ x, int F) {
  int a = blockIdx.x;
  int tid = threadIdx.x;
  int n = ZB * F;
  __shared__ float s1[256], s2[256];
  float s = 0.0f, ss = 0.0f;
  for (int i = tid; i < n; i += 256) {
    int zz = i / F, f = i % F;
    float v = x[((long)zz * NATOMS + a) * F + f];
    s += v; ss += v * v;
  }
  s1[tid] = s; s2[tid] = ss;
  __syncthreads();
  for (int o = 128; o > 0; o >>= 1) {
    if (tid < o) { s1[tid] += s1[tid + o]; s2[tid] += s2[tid + o]; }
    __syncthreads();
  }
  float mean = s1[0] / n;
  float var = s2[0] / n - mean * mean;
  float inv = rsqrtf(var + 1e-5f);
  for (int i = tid; i < n; i += 256) {
    int zz = i / F, f = i % F;
    long idx = ((long)zz * NATOMS + a) * F + f;
    float v = (x[idx] - mean) * inv;
    x[idx] = v > 0.0f ? v : 0.2f * v;
  }
}

__global__ void k_head2(const float* __restrict__ y1, const float* __restrict__ W,
                        const float* __restrict__ b, float* __restrict__ y2) {
  int t = blockIdx.x * blockDim.x + threadIdx.x;
  if (t >= ZB * NATOMS * 32) return;
  int n = t & 31;
  int row = t >> 5;
  float acc = b[n];
  const float* xr = y1 + (long)row * 128;
  for (int k = 0; k < 128; ++k) acc += xr[k] * W[k * 32 + n];
  y2[t] = acc;
}

__global__ void k_reduce(const float* __restrict__ y2, const float* __restrict__ mask,
                         float* __restrict__ out) {
  int t = blockIdx.x * blockDim.x + threadIdx.x;
  if (t >= ZB * 32) return;
  int z = t >> 5, j = t & 31;
  float acc = 0.0f;
  for (int a = 0; a < NATOMS; ++a)
    acc += y2[((long)z * NATOMS + a) * 32 + j] * mask[z * NATOMS + a];
  out[t] = acc;
}

// ---------------- host orchestration ----------------

extern "C" void kernel_launch(void* const* d_in, const int* in_sizes, int n_in,
                              void* d_out, int out_size, void* d_ws, size_t ws_size,
                              hipStream_t stream) {
  (void)in_sizes; (void)n_in; (void)out_size; (void)ws_size;
  const float* features = (const float*)d_in[0];
  const float* geometry = (const float*)d_in[1];
  const float* mask     = (const float*)d_in[2];
  const float* W_bio    = (const float*)d_in[3];
  const float* b_bio    = (const float*)d_in[4];
  const float* W_ch     = (const float*)d_in[5];
  const float* b_ch     = (const float*)d_in[6];
  const float* rW1[2] = {(const float*)d_in[7],  (const float*)d_in[12]};
  const float* rb1[2] = {(const float*)d_in[8],  (const float*)d_in[13]};
  const float* rW2[2] = {(const float*)d_in[9],  (const float*)d_in[14]};
  const float* rb2[2] = {(const float*)d_in[10], (const float*)d_in[15]};
  const float* rWo[2] = {(const float*)d_in[11], (const float*)d_in[16]};
  const float* fW1 = (const float*)d_in[17];
  const float* fb1 = (const float*)d_in[18];
  const float* fW2 = (const float*)d_in[19];
  const float* fb2 = (const float*)d_in[20];
  float* out = (float*)d_out;

  char* ws = (char*)d_ws;
  size_t off = 0;
  auto alloc = [&](size_t bytes) {
    size_t o = off; off += (bytes + 255) & ~(size_t)255; return o;
  };
  unsigned short* basis = (unsigned short*)(ws + alloc((size_t)ZB * NATOMS * NATOMS * KPAD * 2));
  unsigned short* hbuf  = (unsigned short*)(ws + alloc((size_t)ZB * NATOMS * NATOMS * HDIM * 2));
  unsigned short* tmpT  = (unsigned short*)(ws + alloc((size_t)ZB * HDIM * NATOMS * HDIM * 2));
  unsigned short* W1t   = (unsigned short*)(ws + alloc(64 * 64 * 2));
  unsigned short* W2t   = (unsigned short*)(ws + alloc(64 * 64 * 2));
  unsigned short* Wot   = (unsigned short*)(ws + alloc(4096 * 64 * 2));
  unsigned short* fm    = (unsigned short*)(ws + alloc((size_t)ZB * NATOMS * HDIM * 2));
  float* accbuf = (float*)(ws + alloc((size_t)ZB * NATOMS * HDIM * 4));
  float* fstream0 = (float*)(ws + alloc((size_t)ZB * NATOMS * HDIM * 4));
  float* fstream1 = (float*)(ws + alloc((size_t)ZB * NATOMS * HDIM * 4));
  float* fstream[2] = {fstream0, fstream1};
  float* y1 = (float*)(ws + alloc((size_t)ZB * NATOMS * 128 * 4));
  float* y2 = (float*)(ws + alloc((size_t)ZB * NATOMS * 32 * 4));

  const int NF = ZB * NATOMS * HDIM;              // 196608

  k_encode<<<(NF + 255) / 256, 256, 0, stream>>>(
      features, W_bio, b_bio, W_ch, b_ch, fstream[0], fstream[1]);
  k_basis<<<(ZB * NATOMS * NATOMS + 255) / 256, 256, 0, stream>>>(geometry, basis);

  const long MBIG = (long)ZB * NATOMS * NATOMS;   // 589824 rows
  for (int l = 0; l < 2; ++l) {
    k_packT<<<(64 * 64 + 255) / 256, 256, 0, stream>>>(rW1[l], W1t, 40, 64);
    k_packT<<<(64 * 64 + 255) / 256, 256, 0, stream>>>(rW2[l], W2t, 64, 64);
    k_cast_bf<<<(4096 * 64 + 255) / 256, 256, 0, stream>>>(rWo[l], Wot, 4096 * 64);

    // fused G1+G2: hbuf = ssp(ssp(basis@W1+b1)@W2+b2), h1 stays in LDS
    k_h_fused<<<dim3(1, (unsigned)(MBIG / 128), 1), 256, 0, stream>>>(
        basis, W1t, rb1[l], W2t, rb2[l], hbuf);

    for (int s = 0; s < 2; ++s) {
      k_prep_fm<<<(NF + 255) / 256, 256, 0, stream>>>(fstream[s], mask, fm);
      // G3: tmpT[z][j][b*64+h] = sum_i rWo[h][j][i]*fm[z][b][i]
      k_gemm2<<<dim3(4096 / 64, ZB * NATOMS / 128, 1), 256, 0, stream>>>(
          fm, Wot, tmpT, 64, 64, 0, 0, 1, 1);
      // G4: acc[z][a][j] = sum_{b,h} hbuf[z][a][b*64+h]*tmpT[z][j][b*64+h]
      //     K=12288 split 4 ways across blocks, atomic f32 accumulation
      k_zero<<<(NF + 255) / 256, 256, 0, stream>>>(accbuf, NF);
      k_gemm2<<<dim3(1, NATOMS / 64, ZB * KS), 128, 0, stream>>>(
          hbuf, tmpT, accbuf, 12288, 12288 / KS,
          (long)NATOMS * 12288, (long)64 * 12288, KS, 3);
      // f = sp(acc) * mask
      k_finalize<<<(NF + 255) / 256, 256, 0, stream>>>(accbuf, mask, fstream[s]);
    }
  }

  k_head1<<<(ZB * NATOMS * 128 + 255) / 256, 256, 0, stream>>>(
      fstream[0], fstream[1], fW1, fb1, y1);
  k_bn_lrelu<<<NATOMS, 256, 0, stream>>>(y1, 128);
  k_head2<<<(ZB * NATOMS * 32 + 255) / 256, 256, 0, stream>>>(y1, fW2, fb2, y2);
  k_bn_lrelu<<<NATOMS, 256, 0, stream>>>(y2, 32);
  k_reduce<<<(ZB * 32 + 255) / 256, 256, 0, stream>>>(y2, mask, out);
}
